// Conv2D_lowprecision_5128190951901
// MI455X (gfx1250) — 8-bit matrix operand order repaired and checked by reading, compiled, not run on hardware
//
#include <hip/hip_runtime.h>
#include <stdint.h>


typedef __attribute__((ext_vector_type(16))) int   v16i;
typedef __attribute__((ext_vector_type(8)))  float v8f;

#define IMGS 32
#define H    56
#define W    56
#define CIN  256
#define COUT 256
#define TAPS 9
#define KTOT (TAPS * CIN)
#define TH   8
#define TW   8
#define NT   128
#define HALO ((TH + 2) * (TW + 2) * CIN)

#define ACT_TBL 0x44403800u
#define WGT_TBL 0x4438B8C4u

__device__ __forceinline__ void async_copy_b128(uint32_t lds_addr, const void* gptr) {
    asm volatile("global_load_async_to_lds_b128 %0, %1, off"
                 :: "v"(lds_addr), "v"(gptr) : "memory");
}

__global__ void q2_init_kernel(unsigned int* maxbits) {
    if (threadIdx.x == 0 && blockIdx.x == 0) *maxbits = 0u;
}

__global__ void q2_wmax_kernel(const float* __restrict__ w,
                               unsigned int* __restrict__ maxbits, int n) {
    __shared__ unsigned int red[256];
    unsigned int best = 0u;
    for (int i = blockIdx.x * blockDim.x + threadIdx.x; i < n;
         i += gridDim.x * blockDim.x) {
        unsigned int b = __float_as_uint(fabsf(tanhf(w[i])));
        best = (b > best) ? b : best;
    }
    red[threadIdx.x] = best;
    __syncthreads();
    for (int s = 128; s > 0; s >>= 1) {
        if ((int)threadIdx.x < s) {
            unsigned int o = red[threadIdx.x + s];
            if (o > red[threadIdx.x]) red[threadIdx.x] = o;
        }
        __syncthreads();
    }
    if (threadIdx.x == 0) atomicMax(maxbits, red[0]);
}

__global__ void q2_wquant_kernel(const float* __restrict__ w,
                                 const unsigned int* __restrict__ maxbits,
                                 uint8_t* __restrict__ wq_t) {
    int i = blockIdx.x * blockDim.x + threadIdx.x;
    if (i >= TAPS * CIN * COUT) return;
    int co  = i & 255;
    int ci  = (i >> 8) & 255;
    int tap = i >> 16;
    float maxt = __uint_as_float(*maxbits);
    float tn = tanhf(w[i]) / maxt * 0.5f + 0.5f;
    int   q  = (int)__builtin_rintf(tn * 3.0f);
    wq_t[(size_t)co * KTOT + tap * CIN + ci] =
        (uint8_t)((WGT_TBL >> (q << 3)) & 0xffu);
}

__global__ void __launch_bounds__(256, 2)
q2_conv_wmma_kernel(const float* __restrict__ x,
                    const uint8_t* __restrict__ wq_t,
                    float* __restrict__ out) {
    __shared__ __attribute__((aligned(16))) uint8_t sA[HALO];
    __shared__ __attribute__((aligned(16))) uint8_t sB[2][NT * CIN];

    const int tile  = blockIdx.x;
    const int ty    = tile / (W / TW);
    const int tx    = tile % (W / TW);
    const int img   = blockIdx.y;
    const int nbase = blockIdx.z * NT;

    const int t  = threadIdx.x;
    const int y0 = ty * TH - 1;
    const int x0 = tx * TW - 1;

    const int bcol = t >> 1;
    const uint8_t* const wrow = wq_t + (size_t)(nbase)*KTOT;

    #pragma unroll
    for (int i = 0; i < 8; i++) {
        int idx = t + 256 * i;
        int col = idx >> 4;
        int k16 = (idx & 15) << 4;
        uint32_t lds = (uint32_t)(uintptr_t)(void*)(&sB[0][col * CIN + k16]);
        async_copy_b128(lds, wrow + (size_t)col * KTOT + k16);
    }

    #pragma unroll
    for (int i = 0; i < 25; i++) {
        int d   = t + 256 * i;
        int pix = d >> 6;
        int cin = (d & 63) << 2;
        int iy  = pix / (TW + 2);
        int ix  = pix % (TW + 2);
        int gy  = y0 + iy;
        int gx  = x0 + ix;
        uint32_t packed = 0u;
        if (gy >= 0 && gy < H && gx >= 0 && gx < W) {
            const float4 v = *(const float4*)(x + (((size_t)img * H + gy) * W + gx) * CIN + cin);
            uint32_t q0 = (uint32_t)(int)__builtin_rintf(fminf(fmaxf(v.x, 0.f), 1.f) * 3.0f);
            uint32_t q1 = (uint32_t)(int)__builtin_rintf(fminf(fmaxf(v.y, 0.f), 1.f) * 3.0f);
            uint32_t q2 = (uint32_t)(int)__builtin_rintf(fminf(fmaxf(v.z, 0.f), 1.f) * 3.0f);
            uint32_t q3 = (uint32_t)(int)__builtin_rintf(fminf(fmaxf(v.w, 0.f), 1.f) * 3.0f);
            packed =  ((ACT_TBL >> (q0 << 3)) & 0xffu)
                   | (((ACT_TBL >> (q1 << 3)) & 0xffu) << 8)
                   | (((ACT_TBL >> (q2 << 3)) & 0xffu) << 16)
                   | (((ACT_TBL >> (q3 << 3)) & 0xffu) << 24);
        }
        *(uint32_t*)(sA + (d << 2)) = packed;
    }

    const int lane = t & 31;
    const int wave = t >> 5;
    const int wm   = wave >> 2;
    const int wn   = wave & 3;
    const int lrow = lane & 15;
    const int lg   = lane >> 4;

    int pb[2];
    #pragma unroll
    for (int mt = 0; mt < 2; mt++) {
        int m = wm * 32 + mt * 16 + lrow;
        pb[mt] = (((m >> 3) * (TW + 2)) + (m & 7)) * CIN + (lg << 4);
    }
    int bb[2];
    #pragma unroll
    for (int nt = 0; nt < 2; nt++) {
        int n = wn * 32 + nt * 16 + lrow;
        bb[nt] = n * CIN + (lg << 4);
    }

    v8f acc[2][2];
    #pragma unroll
    for (int mt = 0; mt < 2; mt++)
        #pragma unroll
        for (int nt = 0; nt < 2; nt++)
            #pragma unroll
            for (int j = 0; j < 8; j++)
                acc[mt][nt][j] = 0.0f;

    #pragma unroll 1
    for (int tap = 0; tap < TAPS; tap++) {
        const int buf = tap & 1;

        if (tap + 1 < TAPS) {
            const uint8_t* wsrc = wrow + (size_t)(tap + 1) * CIN;
            #pragma unroll
            for (int i = 0; i < 8; i++) {
                int idx = t + 256 * i;
                int col = idx >> 4;
                int k16 = (idx & 15) << 4;
                uint32_t lds = (uint32_t)(uintptr_t)(void*)(&sB[buf ^ 1][col * CIN + k16]);
                async_copy_b128(lds, wsrc + (size_t)col * KTOT + k16);
            }
            asm volatile("s_wait_asynccnt 0x8" ::: "memory");
        } else {
            asm volatile("s_wait_asynccnt 0x0" ::: "memory");
        }
        __syncthreads();

        const int kh = tap / 3;
        const int kw = tap % 3;
        const int toff = (kh * (TW + 2) + kw) * CIN;

        #pragma unroll
        for (int c = 0; c < 2; c++) {
            v16i af[2];
            #pragma unroll
            for (int mt = 0; mt < 2; mt++) {
                const uint8_t* ap = sA + pb[mt] + toff + (c << 7);
                #pragma unroll
                for (int h = 0; h < 2; h++) {
                    #pragma unroll
                    for (int p = 0; p < 4; p++) {
                        uint2 dv = *(const uint2*)(ap + (h << 6) + ((p >> 1) << 5) + ((p & 1) << 3));
                        af[mt][h * 8 + p * 2 + 0] = (int)dv.x;
                        af[mt][h * 8 + p * 2 + 1] = (int)dv.y;
                    }
                }
            }
            v16i bf[2];
            #pragma unroll
            for (int nt = 0; nt < 2; nt++) {
                const uint8_t* bp = &sB[buf][0] + bb[nt] + (c << 7);
                #pragma unroll
                for (int p = 0; p < 4; p++) {
                    uint4 qv = *(const uint4*)(bp + (p << 5));
                    bf[nt][p * 4 + 0] = (int)qv.x;
                    bf[nt][p * 4 + 1] = (int)qv.y;
                    bf[nt][p * 4 + 2] = (int)qv.z;
                    bf[nt][p * 4 + 3] = (int)qv.w;
                }
            }
            #pragma unroll
            for (int mt = 0; mt < 2; mt++)
                #pragma unroll
                for (int nt = 0; nt < 2; nt++)
                    acc[mt][nt] = __builtin_amdgcn_wmma_f32_16x16x128_fp8_fp8(
                        af[mt], bf[nt], (short)0, acc[mt][nt], false, false);
        }
        __syncthreads();
    }

    const float scale = 1.0f / 9.0f;
    #pragma unroll
    for (int mt = 0; mt < 2; mt++) {
        #pragma unroll
        for (int nt = 0; nt < 2; nt++) {
            int n = nbase + wn * 32 + nt * 16 + lrow;
            #pragma unroll
            for (int r = 0; r < 8; r++) {
                int m  = wm * 32 + mt * 16 + r + (lg << 3);
                int gy = ty * TH + (m >> 3);
                int gx = tx * TW + (m & 7);
                out[(((size_t)img * H + gy) * W + gx) * COUT + n] =
                    scale * acc[mt][nt][r];
            }
        }
    }
    (void)bcol;
}

extern "C" void kernel_launch(void* const* d_in, const int* in_sizes, int n_in,
                              void* d_out, int out_size, void* d_ws, size_t ws_size,
                              hipStream_t stream) {
    const float* x = (const float*)d_in[0];
    const float* w = (const float*)d_in[1];
    float* out = (float*)d_out;

    unsigned int* maxbits = (unsigned int*)d_ws;
    uint8_t* wq_t = (uint8_t*)d_ws + 256;

    const int nw = TAPS * CIN * COUT;

    q2_init_kernel<<<1, 1, 0, stream>>>(maxbits);
    q2_wmax_kernel<<<256, 256, 0, stream>>>(w, maxbits, nw);
    q2_wquant_kernel<<<(nw + 255) / 256, 256, 0, stream>>>(w, maxbits, wq_t);

    dim3 grid((H / TH) * (W / TW), IMGS, COUT / NT);
    q2_conv_wmma_kernel<<<grid, 256, 0, stream>>>(x, wq_t, out);
}
